// MultiHeadAttention_86449101734166
// MI455X (gfx1250) — matrix operand order repaired and checked by reading, not compiled, not run on hardware
//
#include <hip/hip_runtime.h>

// ---------------------------------------------------------------------------
// Problem constants (must match reference)
// ---------------------------------------------------------------------------
#define B_  4
#define S_  2048
#define D_  1024
#define H_  16
#define DK_ 64
#define M_  (B_ * S_)          // 8192 token rows

typedef _Float16 h16;
typedef __attribute__((ext_vector_type(16))) _Float16 v16h;
typedef __attribute__((ext_vector_type(8)))  _Float16 v8h;
typedef __attribute__((ext_vector_type(8)))  float    v8f;

static __device__ __forceinline__ v8h ld8(const h16* p) { return *(const v8h*)p; }

static __device__ __forceinline__ v16h cat(v8h lo, v8h hi) {
  v16h r;
#pragma unroll
  for (int i = 0; i < 8; ++i) { r[i] = lo[i]; r[8 + i] = hi[i]; }
  return r;
}

// D = A(16x32 f16) * B(32x16 f16) + C(16x16 f32), wave32
#define WMMA(a, b, c) \
  __builtin_amdgcn_wmma_f32_16x16x32_f16(false, (a), false, (b), (short)0, (c), false, false)

// ---------------------------------------------------------------------------
// fp32 -> f16 conversion (n is a multiple of 4)
// ---------------------------------------------------------------------------
__global__ __launch_bounds__(256) void cvt_kernel(const float* __restrict__ s,
                                                  h16* __restrict__ d, int n) {
  int i = (blockIdx.x * blockDim.x + threadIdx.x) * 4;
  if (i + 3 < n) {
    float4 v = *(const float4*)(s + i);
    d[i + 0] = (h16)v.x;
    d[i + 1] = (h16)v.y;
    d[i + 2] = (h16)v.z;
    d[i + 3] = (h16)v.w;
  }
}

// ---------------------------------------------------------------------------
// Projection GEMM: Y[m,n] = sum_k X[m,k] * W[n,k] + bias[n]
//   one wave -> 32x64 output tile (2 A-frags x 4 B-frags, 8 WMMAs per k-step)
//   vmode 0: write f16 out[b,h,s,dk]  (Q, K)
//   vmode 1: write f16 out[b,h,dk,s]  (V transposed for the PV B-fragment)
// grid = (D_/64, M_/(32*8)), block = 256 (8 waves)
// ---------------------------------------------------------------------------
__global__ __launch_bounds__(256) void proj_kernel(const h16* __restrict__ X,
                                                   const h16* __restrict__ W,
                                                   const float* __restrict__ bias,
                                                   h16* __restrict__ out, int vmode) {
  const int lane  = threadIdx.x & 31;
  const int wave  = threadIdx.x >> 5;
  const int mtile = blockIdx.y * 8 + wave;   // 0..255 (32-row tiles)
  const int n0    = blockIdx.x * 64;
  const int mrow  = mtile * 32;
  const int nl    = lane & 15;
  const int hi    = lane >> 4;
  const int koA   = hi ? 8 : 0;    // A-frag K base (ISA 16-bit A layout)
  const int koB   = hi ? 8 : 0;    // B-frag K base

  const h16* arow0 = X + (size_t)(mrow + nl) * D_ + koA;
  const h16* arow1 = X + (size_t)(mrow + 16 + nl) * D_ + koA;
  const h16* brow0 = W + (size_t)(n0 +  0 + nl) * D_ + koB;
  const h16* brow1 = W + (size_t)(n0 + 16 + nl) * D_ + koB;
  const h16* brow2 = W + (size_t)(n0 + 32 + nl) * D_ + koB;
  const h16* brow3 = W + (size_t)(n0 + 48 + nl) * D_ + koB;

  v8f acc[2][4] = {{v8f{}, v8f{}, v8f{}, v8f{}}, {v8f{}, v8f{}, v8f{}, v8f{}}};
  for (int k = 0; k < D_; k += 32) {
    v16h a0 = cat(ld8(arow0 + k), ld8(arow0 + k + 16));
    v16h a1 = cat(ld8(arow1 + k), ld8(arow1 + k + 16));
    v16h b0 = cat(ld8(brow0 + k), ld8(brow0 + k + 16));
    v16h b1 = cat(ld8(brow1 + k), ld8(brow1 + k + 16));
    v16h b2 = cat(ld8(brow2 + k), ld8(brow2 + k + 16));
    v16h b3 = cat(ld8(brow3 + k), ld8(brow3 + k + 16));
    acc[0][0] = WMMA(a0, b0, acc[0][0]);
    acc[0][1] = WMMA(a0, b1, acc[0][1]);
    acc[0][2] = WMMA(a0, b2, acc[0][2]);
    acc[0][3] = WMMA(a0, b3, acc[0][3]);
    acc[1][0] = WMMA(a1, b0, acc[1][0]);
    acc[1][1] = WMMA(a1, b1, acc[1][1]);
    acc[1][2] = WMMA(a1, b2, acc[1][2]);
    acc[1][3] = WMMA(a1, b3, acc[1][3]);
  }

#pragma unroll
  for (int ms = 0; ms < 2; ++ms) {
#pragma unroll
    for (int t = 0; t < 4; ++t) {
      const int ncol = n0 + t * 16 + nl;
      const float bvv = bias[ncol];
      const int h  = ncol >> 6;        // / DK_
      const int dk = ncol & 63;
#pragma unroll
      for (int r = 0; r < 8; ++r) {
        const int m  = mrow + ms * 16 + r + 8 * hi;   // C-layout row
        const int bb = m >> 11;                       // / S_
        const int ss = m & (S_ - 1);
        const float v = acc[ms][t][r] + bvv;
        size_t idx;
        if (vmode == 0)
          idx = ((size_t)((bb * H_ + h) * S_) + ss) * DK_ + dk;
        else
          idx = ((size_t)((bb * H_ + h) * DK_) + dk) * S_ + ss;
        out[idx] = (h16)v;
      }
    }
  }
}

// ---------------------------------------------------------------------------
// Streaming (flash-style) attention: one wave owns TWO (b, h, 16-query) tiles
// so every K/V fragment load is shared across both tiles (halves K/V traffic).
//   scores  : 8x v_wmma per 32-key chunk (A = Q frags, B = K rows, contiguous)
//   softmax : online max/sum, cross-lane reduce within 16-lane halves
//   P transpose: LDS staging (C-layout -> A-fragment layout), 2 KB per wave
//   PV      : 8x v_wmma per chunk against transposed V (contiguous loads)
// grid = B_*H_*(S_/32)/8 blocks, block = 256 (8 waves)
// ---------------------------------------------------------------------------
__global__ __launch_bounds__(256) void attn_kernel(const h16* __restrict__ Q,
                                                   const h16* __restrict__ K,
                                                   const h16* __restrict__ Vt,
                                                   h16* __restrict__ O) {
  __shared__ h16 psmem[8 * 2 * 512];   // 8 waves * 2 q-tiles * 16x32 f16 P tile
  const int lane = threadIdx.x & 31;
  const int wave = threadIdx.x >> 5;
  const int wg   = blockIdx.x * 8 + wave;   // 0..4095
  const int bh   = wg >> 6;                 // / (S_/32)
  const int qp   = wg & 63;                 // q-tile pair index
  const int nl   = lane & 15;
  const int hi   = lane >> 4;
  const int koA  = hi ? 8 : 0;
  const int koB  = hi ? 8 : 0;
  h16* pbuf = psmem + wave * 1024;

  const h16* Qh = Q  + (size_t)bh * S_ * DK_;
  const h16* Kh = K  + (size_t)bh * S_ * DK_;
  const h16* Vh = Vt + (size_t)bh * DK_ * S_;

  // Q fragments (d = 0..31 and 32..63) for both q-tiles, loaded once
  v16h qlo[2], qhi[2];
#pragma unroll
  for (int qi = 0; qi < 2; ++qi) {
    const h16* qrow = Qh + (size_t)((qp * 2 + qi) * 16 + nl) * DK_;
    qlo[qi] = cat(ld8(qrow + koA),      ld8(qrow + koA + 16));
    qhi[qi] = cat(ld8(qrow + 32 + koA), ld8(qrow + 32 + koA + 16));
  }

  v8f acc[2][4] = {{v8f{}, v8f{}, v8f{}, v8f{}}, {v8f{}, v8f{}, v8f{}, v8f{}}};
  float mrun[2][8], lrun[2][8];
#pragma unroll
  for (int qi = 0; qi < 2; ++qi)
#pragma unroll
    for (int r = 0; r < 8; ++r) { mrun[qi][r] = -1e30f; lrun[qi][r] = 0.0f; }
  const float scale = 0.125f;   // 1/sqrt(DK)

  for (int kb0 = 0; kb0 < S_; kb0 += 32) {
    // ---- K fragments for this 32-key chunk (shared by both q-tiles) ----
    v16h kf[2][2];   // [key tile t][d chunk]
#pragma unroll
    for (int t = 0; t < 2; ++t) {
      const h16* krow = Kh + (size_t)(kb0 + t * 16 + nl) * DK_;
      kf[t][0] = cat(ld8(krow + koB),      ld8(krow + koB + 16));
      kf[t][1] = cat(ld8(krow + 32 + koB), ld8(krow + 32 + koB + 16));
    }

#pragma unroll
    for (int qi = 0; qi < 2; ++qi) {
      // ---- scores for the two 16-key tiles ----
      v8f sc[2];
#pragma unroll
      for (int t = 0; t < 2; ++t) {
        v8f s = {};
        s = WMMA(qlo[qi], kf[t][0], s);
        s = WMMA(qhi[qi], kf[t][1], s);
        sc[t] = s;
      }

      // ---- online softmax; lanes 0-15 own q-rows 0-7, lanes 16-31 rows 8-15
#pragma unroll
      for (int r = 0; r < 8; ++r) {
        float a0 = sc[0][r] * scale;
        float a1 = sc[1][r] * scale;
        float mx = fmaxf(a0, a1);
        mx = fmaxf(mx, __shfl_xor(mx, 1));
        mx = fmaxf(mx, __shfl_xor(mx, 2));
        mx = fmaxf(mx, __shfl_xor(mx, 4));
        mx = fmaxf(mx, __shfl_xor(mx, 8));
        float nm   = fmaxf(mrun[qi][r], mx);
        float corr = __expf(mrun[qi][r] - nm);
        mrun[qi][r] = nm;
        float p0 = __expf(a0 - nm);
        float p1 = __expf(a1 - nm);
        float ps = p0 + p1;
        ps += __shfl_xor(ps, 1);
        ps += __shfl_xor(ps, 2);
        ps += __shfl_xor(ps, 4);
        ps += __shfl_xor(ps, 8);
        lrun[qi][r] = lrun[qi][r] * corr + ps;
        acc[qi][0][r] *= corr; acc[qi][1][r] *= corr;
        acc[qi][2][r] *= corr; acc[qi][3][r] *= corr;
        const int row = r + 8 * hi;
        pbuf[qi * 512 + row * 32 + nl]      = (h16)p0;   // key cols 0-15
        pbuf[qi * 512 + row * 32 + 16 + nl] = (h16)p1;   // key cols 16-31
      }
    }

    // wave-private LDS transpose: C-layout P tiles -> 16x32 A fragments
    __builtin_amdgcn_wave_barrier();
    asm volatile("" ::: "memory");
    v16h pf[2];
#pragma unroll
    for (int qi = 0; qi < 2; ++qi) {
      const h16* prow = pbuf + qi * 512 + nl * 32 + koA;
      pf[qi] = cat(ld8(prow), ld8(prow + 16));
    }
    asm volatile("" ::: "memory");

    // ---- PV accumulation against transposed V (V frag shared by q-tiles) --
#pragma unroll
    for (int dt = 0; dt < 4; ++dt) {
      const h16* vrow = Vh + (size_t)(dt * 16 + nl) * S_ + kb0 + koB;
      v16h vf = cat(ld8(vrow), ld8(vrow + 16));
      acc[0][dt] = WMMA(pf[0], vf, acc[0][dt]);
      acc[1][dt] = WMMA(pf[1], vf, acc[1][dt]);
    }
    __builtin_amdgcn_wave_barrier();
  }

  // ---- normalize and write attn output as [B,S,D] f16 ----
  const int b = bh >> 4;   // / H_
  const int h = bh & 15;
#pragma unroll
  for (int qi = 0; qi < 2; ++qi) {
#pragma unroll
    for (int dt = 0; dt < 4; ++dt) {
#pragma unroll
      for (int r = 0; r < 8; ++r) {
        const int srow = (qp * 2 + qi) * 16 + r + 8 * hi;
        const int col  = h * DK_ + dt * 16 + nl;
        O[((size_t)(b * S_ + srow)) * D_ + col] = (h16)(acc[qi][dt][r] / lrun[qi][r]);
      }
    }
  }
}

// ---------------------------------------------------------------------------
// Final projection: out[m,n] = sum_k A[m,k]*Wo[n,k] + bo[n], fp32 result
//   same 32x64 per-wave tiling as proj_kernel
// ---------------------------------------------------------------------------
__global__ __launch_bounds__(256) void final_kernel(const h16* __restrict__ X,
                                                    const h16* __restrict__ W,
                                                    const float* __restrict__ bias,
                                                    float* __restrict__ out) {
  const int lane  = threadIdx.x & 31;
  const int wave  = threadIdx.x >> 5;
  const int mtile = blockIdx.y * 8 + wave;
  const int n0    = blockIdx.x * 64;
  const int mrow  = mtile * 32;
  const int nl    = lane & 15;
  const int hi    = lane >> 4;
  const int koA   = hi ? 8 : 0;
  const int koB   = hi ? 8 : 0;

  const h16* arow0 = X + (size_t)(mrow + nl) * D_ + koA;
  const h16* arow1 = X + (size_t)(mrow + 16 + nl) * D_ + koA;
  const h16* brow0 = W + (size_t)(n0 +  0 + nl) * D_ + koB;
  const h16* brow1 = W + (size_t)(n0 + 16 + nl) * D_ + koB;
  const h16* brow2 = W + (size_t)(n0 + 32 + nl) * D_ + koB;
  const h16* brow3 = W + (size_t)(n0 + 48 + nl) * D_ + koB;

  v8f acc[2][4] = {{v8f{}, v8f{}, v8f{}, v8f{}}, {v8f{}, v8f{}, v8f{}, v8f{}}};
  for (int k = 0; k < D_; k += 32) {
    v16h a0 = cat(ld8(arow0 + k), ld8(arow0 + k + 16));
    v16h a1 = cat(ld8(arow1 + k), ld8(arow1 + k + 16));
    v16h b0 = cat(ld8(brow0 + k), ld8(brow0 + k + 16));
    v16h b1 = cat(ld8(brow1 + k), ld8(brow1 + k + 16));
    v16h b2 = cat(ld8(brow2 + k), ld8(brow2 + k + 16));
    v16h b3 = cat(ld8(brow3 + k), ld8(brow3 + k + 16));
    acc[0][0] = WMMA(a0, b0, acc[0][0]);
    acc[0][1] = WMMA(a0, b1, acc[0][1]);
    acc[0][2] = WMMA(a0, b2, acc[0][2]);
    acc[0][3] = WMMA(a0, b3, acc[0][3]);
    acc[1][0] = WMMA(a1, b0, acc[1][0]);
    acc[1][1] = WMMA(a1, b1, acc[1][1]);
    acc[1][2] = WMMA(a1, b2, acc[1][2]);
    acc[1][3] = WMMA(a1, b3, acc[1][3]);
  }
#pragma unroll
  for (int ms = 0; ms < 2; ++ms) {
#pragma unroll
    for (int t = 0; t < 4; ++t) {
      const int ncol = n0 + t * 16 + nl;
      const float bvv = bias[ncol];
#pragma unroll
      for (int r = 0; r < 8; ++r) {
        const int m = mrow + ms * 16 + r + 8 * hi;
        out[(size_t)m * D_ + ncol] = acc[ms][t][r] + bvv;
      }
    }
  }
}

// ---------------------------------------------------------------------------
// Host launcher
// ---------------------------------------------------------------------------
extern "C" void kernel_launch(void* const* d_in, const int* in_sizes, int n_in,
                              void* d_out, int out_size, void* d_ws, size_t ws_size,
                              hipStream_t stream) {
  const float* xQ = (const float*)d_in[0];
  const float* xK = (const float*)d_in[1];
  const float* xV = (const float*)d_in[2];
  const float* Wq = (const float*)d_in[3];
  const float* bq = (const float*)d_in[4];
  const float* Wk = (const float*)d_in[5];
  const float* bk = (const float*)d_in[6];
  const float* Wv = (const float*)d_in[7];
  const float* bv = (const float*)d_in[8];
  const float* Wo = (const float*)d_in[9];
  const float* bo = (const float*)d_in[10];

  // Workspace layout (~126 MB total), everything 256B-aligned by construction.
  char* ws = (char*)d_ws;
  const size_t XB = (size_t)M_ * D_ * sizeof(h16);   // 16.78 MB
  const size_t WB = (size_t)D_ * D_ * sizeof(h16);   //  2.10 MB
  h16* xq_h = (h16*)ws; ws += XB;
  h16* xk_h = (h16*)ws; ws += XB;
  h16* xv_h = (h16*)ws; ws += XB;
  h16* wq_h = (h16*)ws; ws += WB;
  h16* wk_h = (h16*)ws; ws += WB;
  h16* wv_h = (h16*)ws; ws += WB;
  h16* wo_h = (h16*)ws; ws += WB;
  h16* Qb   = (h16*)ws; ws += XB;   // [B,H,S,DK]
  h16* Kb   = (h16*)ws; ws += XB;   // [B,H,S,DK]
  h16* Vtb  = (h16*)ws; ws += XB;   // [B,H,DK,S]
  h16* Ab   = (h16*)ws; ws += XB;   // [B,S,D] attention output

  const int nX = M_ * D_;       // 8388608
  const int nW = D_ * D_;       // 1048576
  cvt_kernel<<<nX / 1024, 256, 0, stream>>>(xQ, xq_h, nX);
  cvt_kernel<<<nX / 1024, 256, 0, stream>>>(xK, xk_h, nX);
  cvt_kernel<<<nX / 1024, 256, 0, stream>>>(xV, xv_h, nX);
  cvt_kernel<<<nW / 1024, 256, 0, stream>>>(Wq, wq_h, nW);
  cvt_kernel<<<nW / 1024, 256, 0, stream>>>(Wk, wk_h, nW);
  cvt_kernel<<<nW / 1024, 256, 0, stream>>>(Wv, wv_h, nW);
  cvt_kernel<<<nW / 1024, 256, 0, stream>>>(Wo, wo_h, nW);

  dim3 pg(D_ / 64, M_ / (32 * 8));   // (16, 32)
  proj_kernel<<<pg, 256, 0, stream>>>(xq_h, wq_h, bq, Qb, 0);
  proj_kernel<<<pg, 256, 0, stream>>>(xk_h, wk_h, bk, Kb, 0);
  proj_kernel<<<pg, 256, 0, stream>>>(xv_h, wv_h, bv, Vtb, 1);

  attn_kernel<<<(B_ * H_ * (S_ / 32)) / 8, 256, 0, stream>>>(Qb, Kb, Vtb, Ab);

  final_kernel<<<pg, 256, 0, stream>>>(Ab, wo_h, bo, (float*)d_out);
}
